// CnnAttnLstmCRF_80865644249710
// MI455X (gfx1250) — compile-verified, matrix operand order checked by reading, not run on hardware
//
#include <hip/hip_runtime.h>

typedef __attribute__((ext_vector_type(16))) __bf16 v16bf;
typedef __attribute__((ext_vector_type(8)))  float  v8f;

#define B_    32
#define S_    128
#define DC_   128
#define HC_   200
#define DI_   56
#define DW_   256
#define H_    200
#define G4_   800     // 4*H
#define GCAT_ 1600    // both directions
#define KHH_  224     // 200 padded to mult of 32
#define KH2T_ 416     // 400 padded to mult of 32
#define T_    66
#define NPAD_ 80      // 66 padded to mult of 16
#define START_ 64
#define STOP_  65
#define INVTEMP_ 0.08838834764831845f   // 1/sqrt(128)

// ---------------- WMMA fragment helpers (CDNA5 16x16x32 bf16) ----------------
// A-matrix layout (ISA 7.12.2, 16-bit A 16x32): lane in [0,15] -> M=lane,
// half=lane>>4 selects K ranges; VGPR j packs K = (j>>2)*16 + half*8 + (j&3)*2 + {0,1}.
// B is symmetric with N per-lane. C/D f32: VGPR r -> M = r + 8*half, N = lane&15.

__device__ __forceinline__ v16bf frag_rowK(const __bf16* __restrict__ P, int ld, int r0, int k0) {
  // Loads a fragment where element (row r0+(lane&15), K contiguous in memory).
  // Serves as: A frag for row-major [M,K]; B frag when weights are stored [N,K] row-major.
  v16bf f;
  int lane = threadIdx.x & 31;
  int half = lane >> 4;
  const __bf16* row = P + (size_t)(r0 + (lane & 15)) * ld;
#pragma unroll
  for (int j = 0; j < 8; ++j) {
    int k = k0 + ((j >> 2) << 4) + (half << 3) + ((j & 3) << 1);
    f[2 * j]     = row[k];
    f[2 * j + 1] = row[k + 1];
  }
  return f;
}

__device__ __forceinline__ v16bf frag_colN(const __bf16* __restrict__ P, int ld, int k0, int n0) {
  // B frag when B is row-major [K,N]: B[k][n], n per lane, strided over K.
  v16bf f;
  int lane = threadIdx.x & 31;
  int half = lane >> 4;
  const __bf16* col = P + n0 + (lane & 15);
#pragma unroll
  for (int j = 0; j < 8; ++j) {
    int k = k0 + ((j >> 2) << 4) + (half << 3) + ((j & 3) << 1);
    f[2 * j]     = col[(size_t)k * ld];
    f[2 * j + 1] = col[(size_t)(k + 1) * ld];
  }
  return f;
}

__device__ __forceinline__ v8f wmma_bf16(v16bf a, v16bf b, v8f c) {
  return __builtin_amdgcn_wmma_f32_16x16x32_bf16(false, a, false, b, (short)0, c, false, false);
}

// ---------------- K0: weight conversion / padding to bf16, zero pads ----------------
__global__ __launch_bounds__(256) void k0_prep(
    const float* w_ih_f, const float* w_ih_b,
    const float* b_ih_f, const float* b_hh_f, const float* b_ih_b, const float* b_hh_b,
    const float* w_hh_f, const float* w_hh_b, const float* h2t_w,
    __bf16* wih_cat, float* bias_cat, __bf16* whh_bf, __bf16* h2t_bf, __bf16* lstm_bf)
{
  int tid = blockIdx.x * blockDim.x + threadIdx.x;
  int nt  = gridDim.x * blockDim.x;
  for (int i = tid; i < GCAT_ * DW_; i += nt) {
    int r = i / DW_, k = i % DW_;
    float v = (r < G4_) ? w_ih_f[r * DW_ + k] : w_ih_b[(r - G4_) * DW_ + k];
    wih_cat[i] = (__bf16)v;
  }
  for (int i = tid; i < GCAT_; i += nt)
    bias_cat[i] = (i < G4_) ? (b_ih_f[i] + b_hh_f[i]) : (b_ih_b[i - G4_] + b_hh_b[i - G4_]);
  for (int i = tid; i < 2 * G4_ * KHH_; i += nt) {
    int d = i / (G4_ * KHH_), r = (i / KHH_) % G4_, k = i % KHH_;
    float v = (k < H_) ? (d == 0 ? w_hh_f[r * H_ + k] : w_hh_b[r * H_ + k]) : 0.f;
    whh_bf[i] = (__bf16)v;
  }
  for (int i = tid; i < NPAD_ * KH2T_; i += nt) {
    int r = i / KH2T_, k = i % KH2T_;
    float v = (r < T_ && k < 400) ? h2t_w[r * 400 + k] : 0.f;
    h2t_bf[i] = (__bf16)v;
  }
  // zero K-pad cols (400..415) of lstm_out so K5 A-frags are clean
  for (int i = tid; i < B_ * S_ * 16; i += nt) {
    int r = i / 16, c = 400 + (i % 16);
    lstm_bf[(size_t)r * KH2T_ + c] = (__bf16)0.f;
  }
}

// ---------------- K1: char emb gather + conv1d + intent concat -> q ; wf ----------------
__global__ __launch_bounds__(256) void k1_front(
    const int* bc, const int* bw, const int* bi, const int* bl,
    const float* wind, const float* lind,
    const float* char_emb, const float* word_emb, const float* intent_emb, const float* lexi_emb,
    const float* conv_w, const float* conv_b,
    __bf16* q_bf, __bf16* wf_bf)
{
  __shared__ float ce[3 * DC_];
  int bs = blockIdx.x;             // 0..4095
  int b = bs >> 7, s = bs & 127;
  int tid = threadIdx.x;
  for (int i = tid; i < 3 * DC_; i += 256) {
    int k = i / DC_, c = i % DC_;
    int ss = s + k - 1;
    float v = 0.f;
    if (ss >= 0 && ss < S_) v = char_emb[(size_t)bc[b * S_ + ss] * DC_ + c];
    ce[i] = v;
  }
  __syncthreads();
  if (tid < HC_) {
    float acc = conv_b[tid];
    const float* w = conv_w + (size_t)tid * (DC_ * 3);
#pragma unroll 4
    for (int c = 0; c < DC_; ++c) {
      acc += ce[c]           * w[c * 3 + 0];
      acc += ce[DC_ + c]     * w[c * 3 + 1];
      acc += ce[2 * DC_ + c] * w[c * 3 + 2];
    }
    q_bf[(size_t)bs * DW_ + tid] = (__bf16)acc;
  } else {
    int j = tid - HC_;             // 0..55
    q_bf[(size_t)bs * DW_ + HC_ + j] = (__bf16)intent_emb[bi[b] * DI_ + j];
  }
  {
    float wi = wind[bs], li = lind[bs];
    size_t wrow = (size_t)bw[bs] * DW_, lrow = (size_t)bl[bs] * DW_;
    wf_bf[(size_t)bs * DW_ + tid] = (__bf16)(word_emb[wrow + tid] * wi + lexi_emb[lrow + tid] * li);
  }
}

// ---------------- K2: fused attention per batch (scores -> softmax -> P@wf), all in LDS ----------------
__global__ __launch_bounds__(1024) void k2_attn(
    const __bf16* q_bf, const __bf16* wf_bf, const int* bw, __bf16* attn_bf)
{
  extern __shared__ char smem[];
  float*  sc = (float*)smem;                    // 128*128 f32 scores
  __bf16* pb = (__bf16*)(smem + S_ * S_ * 4);   // 128*128 bf16 probs
  int b = blockIdx.x;
  int tid = threadIdx.x;
  int wave = tid >> 5;
  const __bf16* qb  = q_bf  + (size_t)b * S_ * DW_;
  const __bf16* wfb = wf_bf + (size_t)b * S_ * DW_;
  const int* wb = bw + b * S_;

  // scores = q @ wf^T * (1/TEMP), key-pad mask
  for (int tile = wave; tile < 64; tile += 32) {
    int m0 = (tile >> 3) << 4, n0 = (tile & 7) << 4;
    v8f acc = {};
#pragma unroll
    for (int kk = 0; kk < DW_ / 32; ++kk) {
      v16bf a  = frag_rowK(qb,  DW_, m0, kk * 32);
      v16bf bb = frag_rowK(wfb, DW_, n0, kk * 32);   // B[k][n] = wf[n][k]
      acc = wmma_bf16(a, bb, acc);
    }
    int lane = tid & 31, half = lane >> 4, n = n0 + (lane & 15);
    bool masked = (wb[n] == 0);
#pragma unroll
    for (int r = 0; r < 8; ++r) {
      float v = acc[r] * INVTEMP_;
      if (masked) v = -1.0e30f;
      sc[(m0 + r + 8 * half) * S_ + n] = v;
    }
  }
  __syncthreads();
  // row softmax: 8 lanes per row, 16 cols each; shfl-xor reduce within groups of 8
  {
    int row = tid >> 3, j = tid & 7;
    float* r = sc + row * S_ + j * 16;
    float mx = -3.0e38f;
#pragma unroll
    for (int i = 0; i < 16; ++i) mx = fmaxf(mx, r[i]);
    for (int off = 1; off < 8; off <<= 1) mx = fmaxf(mx, __shfl_xor(mx, off, 8));
    float e[16], sum = 0.f;
#pragma unroll
    for (int i = 0; i < 16; ++i) { e[i] = __expf(r[i] - mx); sum += e[i]; }
    for (int off = 1; off < 8; off <<= 1) sum += __shfl_xor(sum, off, 8);
    float rinv = 1.0f / sum;
    __bf16* pr = pb + row * S_ + j * 16;
#pragma unroll
    for (int i = 0; i < 16; ++i) pr[i] = (__bf16)(e[i] * rinv);
  }
  __syncthreads();
  // attn_out = P @ wf  -> bf16 for the LSTM input GEMM
  for (int tile = wave; tile < 128; tile += 32) {
    int m0 = (tile >> 4) << 4, n0 = (tile & 15) << 4;
    v8f acc = {};
#pragma unroll
    for (int kk = 0; kk < S_ / 32; ++kk) {
      v16bf a  = frag_rowK(pb, S_, m0, kk * 32);
      v16bf bb = frag_colN(wfb, DW_, kk * 32, n0);
      acc = wmma_bf16(a, bb, acc);
    }
    int lane = tid & 31, half = lane >> 4, n = n0 + (lane & 15);
    __bf16* ob = attn_bf + (size_t)b * S_ * DW_;
#pragma unroll
    for (int r = 0; r < 8; ++r) ob[(m0 + r + 8 * half) * DW_ + n] = (__bf16)acc[r];
  }
}

// ---------------- K3: g_pre = attn @ [Wih_f;Wih_b]^T + (b_ih + b_hh)  [4096 x 1600] ----------------
__global__ __launch_bounds__(256) void k3_xgemm(
    const __bf16* attn_bf, const __bf16* wih_cat, const float* bias_cat, float* g_pre)
{
  int tile = blockIdx.x * 8 + (threadIdx.x >> 5);   // 256 x 100 tiles = 25600
  int mt = tile / 100, nt = tile % 100;
  int m0 = mt << 4, n0 = nt << 4;
  v8f acc = {};
#pragma unroll
  for (int kk = 0; kk < DW_ / 32; ++kk) {
    v16bf a  = frag_rowK(attn_bf, DW_, m0, kk * 32);
    v16bf bb = frag_rowK(wih_cat, DW_, n0, kk * 32);
    acc = wmma_bf16(a, bb, acc);
  }
  int lane = threadIdx.x & 31, half = lane >> 4, n = n0 + (lane & 15);
  float bias = bias_cat[n];
#pragma unroll
  for (int r = 0; r < 8; ++r)
    g_pre[(size_t)(m0 + r + 8 * half) * GCAT_ + n] = acc[r] + bias;
}

// ---------------- K4: persistent LSTM recurrence, one workgroup per direction ----------------
__global__ __launch_bounds__(1024) void k4_lstm(
    const float* g_pre, const __bf16* whh_bf, __bf16* lstm_bf)
{
  extern __shared__ char smem[];
  float*  g  = (float*)smem;                               // 32*800 f32 gates
  float*  cc = (float*)(smem + 32 * G4_ * 4);              // 32*200 f32 cell
  __bf16* hb = (__bf16*)(smem + 32 * G4_ * 4 + 32 * H_ * 4); // 32*224 bf16 hidden (K-padded)
  int dir = blockIdx.x;
  int tid = threadIdx.x;
  int wave = tid >> 5;
  const __bf16* whh = whh_bf + (size_t)dir * G4_ * KHH_;

  for (int e = tid; e < 32 * KHH_; e += 1024) hb[e] = (__bf16)0.f;
  for (int e = tid; e < 32 * H_;   e += 1024) cc[e] = 0.f;
  __syncthreads();

  for (int t = 0; t < S_; ++t) {
    int tt = (dir == 0) ? t : (S_ - 1 - t);
    // hW = h @ Whh^T : M=32 (2 tiles) x N=800 (50 tiles), K=224
    for (int tile = wave; tile < 100; tile += 32) {
      int m0 = (tile & 1) << 4, n0 = (tile >> 1) << 4;
      v8f acc = {};
#pragma unroll
      for (int kk = 0; kk < KHH_ / 32; ++kk) {
        v16bf a  = frag_rowK(hb,  KHH_, m0, kk * 32);
        v16bf bb = frag_rowK(whh, KHH_, n0, kk * 32);
        acc = wmma_bf16(a, bb, acc);
      }
      int lane = tid & 31, half = lane >> 4, n = n0 + (lane & 15);
#pragma unroll
      for (int r = 0; r < 8; ++r) g[(m0 + r + 8 * half) * G4_ + n] = acc[r];
    }
    __syncthreads();
    // gates (i,f,g,o), update c/h
    for (int e = tid; e < 32 * H_; e += 1024) {
      int bb2 = e / H_, j = e % H_;
      const float* gp = g_pre + ((size_t)bb2 * S_ + tt) * GCAT_ + dir * G4_;
      float gi = g[bb2 * G4_ + j]          + gp[j];
      float gf = g[bb2 * G4_ + H_ + j]     + gp[H_ + j];
      float gg = g[bb2 * G4_ + 2 * H_ + j] + gp[2 * H_ + j];
      float go = g[bb2 * G4_ + 3 * H_ + j] + gp[3 * H_ + j];
      float si = 1.f / (1.f + __expf(-gi));
      float sf = 1.f / (1.f + __expf(-gf));
      float so = 1.f / (1.f + __expf(-go));
      float cn = sf * cc[e] + si * tanhf(gg);
      float hn = so * tanhf(cn);
      cc[e] = cn;
      hb[bb2 * KHH_ + j] = (__bf16)hn;
      lstm_bf[((size_t)bb2 * S_ + tt) * KH2T_ + dir * H_ + j] = (__bf16)hn;
    }
    __syncthreads();
  }
}

// ---------------- K5: emissions = lstm_out @ h2t_w^T + h2t_b  [4096 x 80(66)] ----------------
__global__ __launch_bounds__(256) void k5_h2t(
    const __bf16* lstm_bf, const __bf16* h2t_bf, const float* h2t_b, float* emis)
{
  int tile = blockIdx.x * 8 + (threadIdx.x >> 5);   // 256 x 5 tiles = 1280
  int mt = tile / 5, nt = tile % 5;
  int m0 = mt << 4, n0 = nt << 4;
  v8f acc = {};
#pragma unroll
  for (int kk = 0; kk < KH2T_ / 32; ++kk) {
    v16bf a  = frag_rowK(lstm_bf, KH2T_, m0, kk * 32);
    v16bf bb = frag_rowK(h2t_bf,  KH2T_, n0, kk * 32);
    acc = wmma_bf16(a, bb, acc);
  }
  int lane = threadIdx.x & 31, half = lane >> 4, n = n0 + (lane & 15);
  float bias = (n < T_) ? h2t_b[n] : 0.f;
#pragma unroll
  for (int r = 0; r < 8; ++r)
    emis[(size_t)(m0 + r + 8 * half) * NPAD_ + n] = acc[r] + bias;
}

// ---------------- K6: CRF forward (NLL pieces) + Viterbi per batch ----------------
__global__ __launch_bounds__(128) void k6_crf(
    const float* emis, const float* trans, const int* labels,
    int* bpbuf, float* lossbuf, float* out)
{
  __shared__ float alpha[T_], delta[T_], na[T_], nd[T_];
  int b = blockIdx.x;
  int j = threadIdx.x;
  const float* em = emis + (size_t)b * S_ * NPAD_;
  int* bp = bpbuf + (size_t)b * S_ * T_;

  if (j < T_) {
    float a0 = trans[START_ * T_ + j] + em[j];
    alpha[j] = a0;
    delta[j] = a0;
  }
  __syncthreads();
  for (int t = 1; t < S_; ++t) {
    if (j < T_) {
      const float* emt = em + t * NPAD_;
      float mx = -3.0e38f;
      for (int i = 0; i < T_; ++i) mx = fmaxf(mx, alpha[i] + trans[i * T_ + j]);
      float sum = 0.f;
      for (int i = 0; i < T_; ++i) sum += __expf(alpha[i] + trans[i * T_ + j] - mx);
      na[j] = mx + __logf(sum) + emt[j];
      float best = -3.0e38f; int arg = 0;
      for (int i = 0; i < T_; ++i) {
        float v = delta[i] + trans[i * T_ + j];
        if (v > best) { best = v; arg = i; }
      }
      nd[j] = best + emt[j];
      bp[t * T_ + j] = arg;
    }
    __syncthreads();
    if (j < T_) { alpha[j] = na[j]; delta[j] = nd[j]; }
    __syncthreads();
  }
  if (j == 0) {
    float mx = -3.0e38f;
    for (int i = 0; i < T_; ++i) mx = fmaxf(mx, alpha[i] + trans[i * T_ + STOP_]);
    float sum = 0.f;
    for (int i = 0; i < T_; ++i) sum += __expf(alpha[i] + trans[i * T_ + STOP_] - mx);
    float logZ = mx + __logf(sum);
    const int* lab = labels + b * S_;
    float gold = trans[START_ * T_ + lab[0]];
    for (int t = 0; t < S_; ++t) gold += em[t * NPAD_ + lab[t]];
    for (int t = 1; t < S_; ++t) gold += trans[lab[t - 1] * T_ + lab[t]];
    gold += trans[lab[S_ - 1] * T_ + STOP_];
    lossbuf[b] = logZ - gold;
    // Viterbi backtrace
    float best = -3.0e38f; int tag = 0;
    for (int i = 0; i < T_; ++i) {
      float v = delta[i] + trans[i * T_ + STOP_];
      if (v > best) { best = v; tag = i; }
    }
    float* seq = out + 1 + b * S_;
    seq[S_ - 1] = (float)tag;
    for (int t = S_ - 1; t >= 1; --t) {
      tag = bp[t * T_ + tag];
      seq[t - 1] = (float)tag;
    }
  }
}

// ---------------- K7: deterministic loss reduction ----------------
__global__ void k7_sum(const float* partial, float* out) {
  if (threadIdx.x == 0 && blockIdx.x == 0) {
    float s = 0.f;
    for (int i = 0; i < B_; ++i) s += partial[i];
    out[0] = s;
  }
}

extern "C" void kernel_launch(void* const* d_in, const int* in_sizes, int n_in,
                              void* d_out, int out_size, void* d_ws, size_t ws_size,
                              hipStream_t stream)
{
  (void)in_sizes; (void)n_in; (void)out_size; (void)ws_size;
  const int*   bc        = (const int*)d_in[0];
  const int*   bw        = (const int*)d_in[1];
  const int*   bi        = (const int*)d_in[2];
  const int*   bl        = (const int*)d_in[3];
  const float* lind      = (const float*)d_in[6];
  const float* wind      = (const float*)d_in[7];
  const int*   lab       = (const int*)d_in[8];
  const float* char_emb  = (const float*)d_in[9];
  const float* word_emb  = (const float*)d_in[10];
  const float* intent_emb= (const float*)d_in[11];
  const float* lexi_emb  = (const float*)d_in[12];
  const float* conv_w    = (const float*)d_in[13];
  const float* conv_b    = (const float*)d_in[14];
  const float* w_ih_f    = (const float*)d_in[15];
  const float* w_hh_f    = (const float*)d_in[16];
  const float* b_ih_f    = (const float*)d_in[17];
  const float* b_hh_f    = (const float*)d_in[18];
  const float* w_ih_b    = (const float*)d_in[19];
  const float* w_hh_b    = (const float*)d_in[20];
  const float* b_ih_b    = (const float*)d_in[21];
  const float* b_hh_b    = (const float*)d_in[22];
  const float* h2t_w     = (const float*)d_in[23];
  const float* h2t_b     = (const float*)d_in[24];
  const float* trans     = (const float*)d_in[25];
  float* out = (float*)d_out;

  char* ws = (char*)d_ws;
  size_t off = 0;
  auto alloc = [&](size_t bytes) { size_t o = off; off += (bytes + 255) & ~(size_t)255; return o; };
  __bf16* q_bf    = (__bf16*)(ws + alloc((size_t)B_ * S_ * DW_ * 2));
  __bf16* wf_bf   = (__bf16*)(ws + alloc((size_t)B_ * S_ * DW_ * 2));
  __bf16* attn_bf = (__bf16*)(ws + alloc((size_t)B_ * S_ * DW_ * 2));
  __bf16* wih_cat = (__bf16*)(ws + alloc((size_t)GCAT_ * DW_ * 2));
  float*  bias_cat= (float*) (ws + alloc((size_t)GCAT_ * 4));
  __bf16* whh_bf  = (__bf16*)(ws + alloc((size_t)2 * G4_ * KHH_ * 2));
  __bf16* h2t_bf  = (__bf16*)(ws + alloc((size_t)NPAD_ * KH2T_ * 2));
  float*  g_pre   = (float*) (ws + alloc((size_t)B_ * S_ * GCAT_ * 4));
  __bf16* lstm_bf = (__bf16*)(ws + alloc((size_t)B_ * S_ * KH2T_ * 2));
  float*  emis    = (float*) (ws + alloc((size_t)B_ * S_ * NPAD_ * 4));
  int*    bpbuf   = (int*)   (ws + alloc((size_t)B_ * S_ * T_ * 4));
  float*  lossbuf = (float*) (ws + alloc((size_t)B_ * 4));

  k0_prep<<<512, 256, 0, stream>>>(w_ih_f, w_ih_b, b_ih_f, b_hh_f, b_ih_b, b_hh_b,
                                   w_hh_f, w_hh_b, h2t_w,
                                   wih_cat, bias_cat, whh_bf, h2t_bf, lstm_bf);
  k1_front<<<B_ * S_, 256, 0, stream>>>(bc, bw, bi, bl, wind, lind,
                                        char_emb, word_emb, intent_emb, lexi_emb,
                                        conv_w, conv_b, q_bf, wf_bf);
  k2_attn<<<B_, 1024, S_ * S_ * 4 + S_ * S_ * 2, stream>>>(q_bf, wf_bf, bw, attn_bf);
  k3_xgemm<<<3200, 256, 0, stream>>>(attn_bf, wih_cat, bias_cat, g_pre);
  k4_lstm<<<2, 1024, 32 * G4_ * 4 + 32 * H_ * 4 + 32 * KHH_ * 2, stream>>>(g_pre, whh_bf, lstm_bf);
  k5_h2t<<<160, 256, 0, stream>>>(lstm_bf, h2t_bf, h2t_b, emis);
  k6_crf<<<B_, 128, 0, stream>>>(emis, trans, lab, bpbuf, lossbuf, out);
  k7_sum<<<1, 32, 0, stream>>>(lossbuf, out);
}
